// CrossFrameAttnProcessor_15324443312496
// MI455X (gfx1250) — matrix operand order repaired and checked by reading, not compiled, not run on hardware
//
#include <hip/hip_runtime.h>

#define BB 8
#define SS 4096
#define TT 77
#define CC 1280
#define CE 2048
#define HH 20
#define HD 1280

#define STRENGTH 0.5f
#define ALPHA 0.7f
#define FIW 0.5f

typedef __attribute__((ext_vector_type(16))) __bf16 v16bf;
typedef __attribute__((ext_vector_type(8)))  float v8f;

union BFrag { unsigned int u[8]; v16bf v; };

#if __has_builtin(__builtin_amdgcn_global_load_async_to_lds_b128)
#define HAS_ASYNC_LDS 1
typedef __attribute__((ext_vector_type(4))) int v4i_t;
typedef __attribute__((address_space(1))) v4i_t* GV4;
typedef __attribute__((address_space(3))) v4i_t* LV4;
#else
#define HAS_ASYNC_LDS 0
#endif

__device__ __forceinline__ void wait_async_lds() {
#if HAS_ASYNC_LDS
#if __has_builtin(__builtin_amdgcn_s_wait_asynccnt)
  __builtin_amdgcn_s_wait_asynccnt(0);
#else
  asm volatile("s_wait_asynccnt 0x0" ::: "memory");
#endif
#endif
}

__device__ __forceinline__ unsigned short f2bf(float x) {
  unsigned int u = __float_as_uint(x);
  u += 0x7FFFu + ((u >> 16) & 1u);          // round-to-nearest-even
  return (unsigned short)(u >> 16);
}

__device__ __forceinline__ v8f wmma_bf16(const BFrag& a, const BFrag& b, v8f c) {
  return __builtin_amdgcn_wmma_f32_16x16x32_bf16(false, a.v, false, b.v,
                                                 (short)0, c, false, false);
}

// A-matrix 16x32 bf16 frag (ISA 7.12.2): lane holds row m=lane&15,
// dword jj -> K pair klo = ((jj&4)<<2) + 8*(lane>>4) + 2*(jj&3)
__device__ __forceinline__ void load_fragA(BFrag& f, const unsigned short* rowptr, int lane) {
  int hi = (lane >> 4) << 3;
#pragma unroll
  for (int jj = 0; jj < 8; ++jj) {
    int klo = ((jj & 4) << 2) + hi + ((jj & 3) << 1);
    f.u[jj] = *(const unsigned int*)(rowptr + klo);
  }
}
// B-matrix 32x16 bf16 frag: lane holds col n=lane&15,
// dword jj -> K pair klo = ((jj&4)<<2) + 8*(lane>>4) + 2*(jj&3)
__device__ __forceinline__ void load_fragB(BFrag& f, const unsigned short* colptr, int lane) {
  int hi = (lane >> 4) << 3;
#pragma unroll
  for (int jj = 0; jj < 8; ++jj) {
    f.u[jj] = *(const unsigned int*)(colptr + ((jj & 4) << 2) + hi + ((jj & 3) << 1));
  }
}

// ---------------- fp32 -> bf16 bulk conversion
__global__ __launch_bounds__(256) void cvt_bf16_kernel(
    const float* __restrict__ x, unsigned short* __restrict__ y, int n)
{
  int i4 = (blockIdx.x * 256 + threadIdx.x) * 4;
  if (i4 >= n) return;
  float4 v = *(const float4*)(x + i4);
  y[i4 + 0] = f2bf(v.x);
  y[i4 + 1] = f2bf(v.y);
  y[i4 + 2] = f2bf(v.z);
  y[i4 + 3] = f2bf(v.w);
}

// ---------------- GEMM: C[M,N] = A[M,K] @ B[K,N], bf16 operands, fp32 acc
// Block tile 128x64, 4 waves x (32x64). EPI 0: bf16 out; 1: f32 out;
// 2: f32 + bias + residual; 3: bf16 + f32 out.
template<int EPI>
__global__ __launch_bounds__(128) void gemm_bf16(
    const unsigned short* __restrict__ A, const unsigned short* __restrict__ Bm,
    int M, int N, int K,
    unsigned short* __restrict__ Obf, float* __restrict__ Of,
    const float* __restrict__ bias, const float* __restrict__ resid)
{
  __shared__ unsigned short sA[128 * 32];  // [m][k]
  __shared__ unsigned short sBT[64 * 32];  // [n][k]
  const int tid = threadIdx.x;
  const int lane = tid & 31;
  const int wave = tid >> 5;
  const int bm = blockIdx.x * 128;
  const int bn = blockIdx.y * 64;
  const int wm = wave * 32;

  v8f acc[2][4] = {};

  for (int k0 = 0; k0 < K; k0 += 32) {
    // stage A: 128x32 bf16 = 512 16-byte chunks (4 per thread)
    for (int i = tid; i < 512; i += 128) {
      int m = i >> 2, q = i & 3;
      int gm = bm + m;
      unsigned short* lp = sA + m * 32 + q * 8;
      if (gm < M) {
        const unsigned short* gp = A + (size_t)gm * K + k0 + q * 8;
#if HAS_ASYNC_LDS
        __builtin_amdgcn_global_load_async_to_lds_b128((GV4)gp, (LV4)lp, 0, 0);
#else
        *(uint4*)lp = *(const uint4*)gp;
#endif
      } else {
        *(uint4*)lp = make_uint4(0u, 0u, 0u, 0u);
      }
    }
    // stage B transposed: 8 cols at one k per chunk, 256 chunks
    for (int i = tid; i < 256; i += 128) {
      int k = i & 31, n0 = (i >> 5) * 8;
      uint4 v = *(const uint4*)(Bm + (size_t)(k0 + k) * N + bn + n0);
      const unsigned short* pv = (const unsigned short*)&v;
#pragma unroll
      for (int j = 0; j < 8; ++j) sBT[(n0 + j) * 32 + k] = pv[j];
    }
    wait_async_lds();
    __syncthreads();
    BFrag a[2], b[4];
    load_fragA(a[0], sA + (wm + (lane & 15)) * 32, lane);
    load_fragA(a[1], sA + (wm + 16 + (lane & 15)) * 32, lane);
#pragma unroll
    for (int j = 0; j < 4; ++j)
      load_fragB(b[j], sBT + (j * 16 + (lane & 15)) * 32, lane);
#pragma unroll
    for (int i = 0; i < 2; ++i)
#pragma unroll
      for (int j = 0; j < 4; ++j)
        acc[i][j] = wmma_bf16(a[i], b[j], acc[i][j]);
    __syncthreads();
  }
  const int hi = lane >> 4;
#pragma unroll
  for (int i = 0; i < 2; ++i)
#pragma unroll
    for (int j = 0; j < 4; ++j)
#pragma unroll
      for (int r = 0; r < 8; ++r) {
        int gm = bm + wm + i * 16 + hi * 8 + r;
        int gn = bn + j * 16 + (lane & 15);
        if (gm < M) {
          float v = acc[i][j][r];
          if (EPI == 0)      Obf[(size_t)gm * N + gn] = f2bf(v);
          else if (EPI == 1) Of[(size_t)gm * N + gn] = v;
          else if (EPI == 2) Of[(size_t)gm * N + gn] =
                               v + bias[gn] + resid[(size_t)gm * N + gn];
          else { Obf[(size_t)gm * N + gn] = f2bf(v);
                 Of[(size_t)gm * N + gn] = v; }
        }
      }
}

// ---------------- shared attention helper (QK^T -> sS, scaled)
__device__ __forceinline__ void qk_scores(
    const unsigned short* __restrict__ qbf, const unsigned short* __restrict__ kbf,
    unsigned short* sK, float* sS, int b, int h, int sblk,
    int tid, int lane, int wave)
{
  // stage K tile [t][d] bf16 (80x64), t>=77 zero-padded
  for (int i = tid; i < 80 * 16; i += 128) {
    int t = i >> 4, q = i & 15;
    uint4 v = make_uint4(0u, 0u, 0u, 0u);
    if (t < TT)
      v = *(const uint4*)(kbf + ((size_t)(b * TT + t)) * HD + h * 64 + q * 4 * 2);
    *(uint4*)(sK + t * 64 + q * 8) = v;
  }
  __syncthreads();
  const int wm = wave * 16;
  v8f acc[5] = {};
  const size_t qrow = ((size_t)b * SS + sblk * 64 + wm + (lane & 15)) * HD + h * 64;
  for (int kk = 0; kk < 64; kk += 32) {
    BFrag a; load_fragA(a, qbf + qrow + kk, lane);
#pragma unroll
    for (int nt = 0; nt < 5; ++nt) {
      BFrag bf; load_fragB(bf, sK + (nt * 16 + (lane & 15)) * 64 + kk, lane);
      acc[nt] = wmma_bf16(a, bf, acc[nt]);
    }
  }
  const int hi = lane >> 4;
#pragma unroll
  for (int nt = 0; nt < 5; ++nt)
#pragma unroll
    for (int r = 0; r < 8; ++r)
      sS[(wm + hi * 8 + r) * 80 + nt * 16 + (lane & 15)] = acc[nt][r] * 0.125f;
  __syncthreads();
}

// ---------------- pass 1: softmax column sums -> tok_scores
__global__ __launch_bounds__(128) void attn_tok_kernel(
    const unsigned short* __restrict__ qbf, const unsigned short* __restrict__ kbf,
    float* __restrict__ tok_scores)
{
  __shared__ unsigned short sK[80 * 64];
  __shared__ float sS[64 * 80];
  const int tid = threadIdx.x, lane = tid & 31, wave = tid >> 5;
  const int sblk = blockIdx.x, h = blockIdx.y, b = blockIdx.z;
  qk_scores(qbf, kbf, sK, sS, b, h, sblk, tid, lane, wave);
  if (tid < 64) {
    float mx = -1e30f;
    for (int t = 0; t < TT; ++t) mx = fmaxf(mx, sS[tid * 80 + t]);
    float den = 0.0f;
    for (int t = 0; t < TT; ++t) { float e = expf(sS[tid * 80 + t] - mx); sS[tid * 80 + t] = e; den += e; }
    float inv = 1.0f / den;
    for (int t = 0; t < TT; ++t) sS[tid * 80 + t] *= inv;
  }
  __syncthreads();
  if (tid < TT) {
    float cs = 0.0f;
    for (int r = 0; r < 64; ++r) cs += sS[r * 80 + tid];
    atomicAdd(&tok_scores[b * TT + tid], cs);
  }
}

// ---------------- pass 1b: masked attention mass -> pos_scores
__global__ __launch_bounds__(128) void attn_pos_kernel(
    const unsigned short* __restrict__ qbf, const unsigned short* __restrict__ kbf,
    const int* __restrict__ tok_mask, float* __restrict__ pos_scores)
{
  __shared__ unsigned short sK[80 * 64];
  __shared__ float sS[64 * 80];
  const int tid = threadIdx.x, lane = tid & 31, wave = tid >> 5;
  const int sblk = blockIdx.x, h = blockIdx.y, b = blockIdx.z;
  qk_scores(qbf, kbf, sK, sS, b, h, sblk, tid, lane, wave);
  if (tid < 64) {
    float mx = -1e30f;
    for (int t = 0; t < TT; ++t) mx = fmaxf(mx, sS[tid * 80 + t]);
    float den = 0.0f, msum = 0.0f;
    for (int t = 0; t < TT; ++t) {
      float e = expf(sS[tid * 80 + t] - mx);
      den += e;
      if (tok_mask[b * TT + t]) msum += e;
    }
    atomicAdd(&pos_scores[b * SS + sblk * 64 + tid], msum / den);
  }
}

// ---------------- pass 2: softmax(QK2^T) @ V2 -> context fp32
__global__ __launch_bounds__(128) void attn2_kernel(
    const unsigned short* __restrict__ qbf, const unsigned short* __restrict__ k2bf,
    const unsigned short* __restrict__ v2bf, float* __restrict__ ctx)
{
  __shared__ unsigned short sK[80 * 64];   // [t][d]
  __shared__ unsigned short sVT[64 * 96];  // [d][t], t padded to 96
  __shared__ float sS[64 * 80];
  __shared__ unsigned short sP[64 * 96];   // probs bf16 [s][t]
  const int tid = threadIdx.x, lane = tid & 31, wave = tid >> 5;
  const int sblk = blockIdx.x, h = blockIdx.y, b = blockIdx.z;
  // stage V^T: 8 d-values at one t per chunk
  for (int i = tid; i < 96 * 8; i += 128) {
    int t = i >> 3, d0 = (i & 7) * 8;
    if (t < TT) {
      uint4 v = *(const uint4*)(v2bf + ((size_t)(b * TT + t)) * HD + h * 64 + d0);
      const unsigned short* pv = (const unsigned short*)&v;
#pragma unroll
      for (int j = 0; j < 8; ++j) sVT[(d0 + j) * 96 + t] = pv[j];
    } else {
#pragma unroll
      for (int j = 0; j < 8; ++j) sVT[(d0 + j) * 96 + t] = 0;
    }
  }
  qk_scores(qbf, k2bf, sK, sS, b, h, sblk, tid, lane, wave);
  if (tid < 64) {
    float mx = -1e30f;
    for (int t = 0; t < TT; ++t) mx = fmaxf(mx, sS[tid * 80 + t]);
    float den = 0.0f;
    for (int t = 0; t < TT; ++t) den += expf(sS[tid * 80 + t] - mx);
    float inv = 1.0f / den;
    for (int t = 0; t < 96; ++t)
      sP[tid * 96 + t] = (t < TT) ? f2bf(expf(sS[tid * 80 + t] - mx) * inv)
                                  : (unsigned short)0;
  }
  __syncthreads();
  const int wm = wave * 16;
  v8f acc2[4] = {};
  for (int kk = 0; kk < 96; kk += 32) {
    BFrag a; load_fragA(a, sP + (wm + (lane & 15)) * 96 + kk, lane);
#pragma unroll
    for (int nt = 0; nt < 4; ++nt) {
      BFrag bf; load_fragB(bf, sVT + (nt * 16 + (lane & 15)) * 96 + kk, lane);
      acc2[nt] = wmma_bf16(a, bf, acc2[nt]);
    }
  }
  const int hi = lane >> 4;
#pragma unroll
  for (int nt = 0; nt < 4; ++nt)
#pragma unroll
    for (int r = 0; r < 8; ++r) {
      size_t row = (size_t)b * SS + sblk * 64 + wm + hi * 8 + r;
      int col = h * 64 + nt * 16 + (lane & 15);
      ctx[row * HD + col] = acc2[nt][r];
    }
}

// ---------------- stable top-k mask via rank counting (jax top_k tie order)
__global__ void topk_mask_kernel(
    const float* __restrict__ scores, int* __restrict__ mask, int L, int k)
{
  __shared__ float sh[4096];
  const int b = blockIdx.x;
  for (int i = threadIdx.x; i < L; i += blockDim.x) sh[i] = scores[b * L + i];
  __syncthreads();
  for (int i = threadIdx.x; i < L; i += blockDim.x) {
    float si = sh[i];
    int rank = 0;
    for (int j = 0; j < L; ++j) {
      float sj = sh[j];
      rank += (sj > si) || (sj == si && j < i);
    }
    mask[b * L + i] = (rank < k) ? 1 : 0;
  }
}

// ---------------- cross-frame K/V mixing on subject tokens -> bf16 K2/V2
__global__ __launch_bounds__(256) void mix_kv_kernel(
    const float* __restrict__ kp, const float* __restrict__ vp,
    const int* __restrict__ tok_mask,
    unsigned short* __restrict__ k2bf, unsigned short* __restrict__ v2bf)
{
  int i = blockIdx.x * 256 + threadIdx.x;
  if (i >= TT * HD) return;
  int t = i / HD, c = i % HD;
  float xk[BB], xv[BB], sk = 0.0f, sv = 0.0f;
#pragma unroll
  for (int b = 0; b < BB; ++b) {
    size_t idx = ((size_t)(b * TT + t)) * HD + c;
    xk[b] = kp[idx]; xv[b] = vp[idx];
    sk += xk[b]; sv += xv[b];
  }
#pragma unroll
  for (int b = 0; b < BB; ++b) {
    size_t idx = ((size_t)(b * TT + t)) * HD + c;
    float tm = tok_mask[b * TT + t] ? STRENGTH : 0.0f;
    float dk = (1.0f - ALPHA) * ((sk - xk[b]) / (float)(BB - 1) - xk[b]);
    float dv = (1.0f - ALPHA) * ((sv - xv[b]) / (float)(BB - 1) - xv[b]);
    k2bf[idx] = f2bf(xk[b] + tm * dk);
    v2bf[idx] = f2bf(xv[b] + tm * dv);
  }
}

// ---------------- patch injection; also emits bf16 context for final GEMM
__global__ __launch_bounds__(256) void inject_kernel(
    const float* __restrict__ ctx, const int* __restrict__ pos_mask,
    unsigned short* __restrict__ ctxbf)
{
  int i = blockIdx.x * 256 + threadIdx.x;
  if (i >= SS * HD) return;
  int s = i / HD, c = i % HD;
  float x[BB], sum = 0.0f;
#pragma unroll
  for (int b = 0; b < BB; ++b) {
    x[b] = ctx[((size_t)(b * SS + s)) * HD + c];
    sum += x[b];
  }
  const float w = FIW * STRENGTH;
#pragma unroll
  for (int b = 0; b < BB; ++b) {
    float v = x[b];
    if (pos_mask[b * SS + s]) {
      float mo = (sum - v) / (float)(BB - 1);
      v = v + w * (mo - v);
    }
    ctxbf[((size_t)(b * SS + s)) * HD + c] = f2bf(v);
  }
}

extern "C" void kernel_launch(void* const* d_in, const int* in_sizes, int n_in,
                              void* d_out, int out_size, void* d_ws, size_t ws_size,
                              hipStream_t stream) {
  (void)in_sizes; (void)n_in; (void)out_size; (void)ws_size;
  const float* hidden = (const float*)d_in[0];
  const float* enc    = (const float*)d_in[1];
  const float* Wq     = (const float*)d_in[2];
  const float* Wk     = (const float*)d_in[3];
  const float* Wv     = (const float*)d_in[4];
  const float* Wo     = (const float*)d_in[5];
  const float* bo     = (const float*)d_in[6];
  float* out = (float*)d_out;

  char* w = (char*)d_ws;
  auto carve = [&](size_t bytes) -> char* {
    char* p = w; w += (bytes + 255) & ~(size_t)255; return p;
  };
  unsigned short* hbf   = (unsigned short*)carve((size_t)BB * SS * CC * 2);
  unsigned short* encbf = (unsigned short*)carve((size_t)BB * TT * CE * 2);
  unsigned short* wqbf  = (unsigned short*)carve((size_t)CC * HD * 2);
  unsigned short* wkbf  = (unsigned short*)carve((size_t)CE * HD * 2);
  unsigned short* wvbf  = (unsigned short*)carve((size_t)CE * HD * 2);
  unsigned short* wobf  = (unsigned short*)carve((size_t)HD * CC * 2);
  unsigned short* qbf   = (unsigned short*)carve((size_t)BB * SS * HD * 2);
  unsigned short* kbf   = (unsigned short*)carve((size_t)BB * TT * HD * 2);
  unsigned short* k2bf  = (unsigned short*)carve((size_t)BB * TT * HD * 2);
  unsigned short* v2bf  = (unsigned short*)carve((size_t)BB * TT * HD * 2);
  unsigned short* ctxbf = (unsigned short*)carve((size_t)BB * SS * HD * 2);
  float* kp  = (float*)carve((size_t)BB * TT * HD * 4);
  float* vp  = (float*)carve((size_t)BB * TT * HD * 4);
  float* ctx = (float*)carve((size_t)BB * SS * HD * 4);
  float* tok_scores = (float*)carve((size_t)BB * TT * 4);
  float* pos_scores = (float*)carve((size_t)BB * SS * 4);
  int* tok_mask = (int*)carve((size_t)BB * TT * 4);
  int* pos_mask = (int*)carve((size_t)BB * SS * 4);

  // bf16 pre-conversion of all GEMM operands
  auto cvt = [&](const float* src, unsigned short* dst, int n) {
    cvt_bf16_kernel<<<(n / 4 + 255) / 256, 256, 0, stream>>>(src, dst, n);
  };
  cvt(hidden, hbf, BB * SS * CC);
  cvt(enc,    encbf, BB * TT * CE);
  cvt(Wq,     wqbf, CC * HD);
  cvt(Wk,     wkbf, CE * HD);
  cvt(Wv,     wvbf, CE * HD);
  cvt(Wo,     wobf, HD * CC);

  // Projections (bf16 WMMA, fp32 accumulate)
  gemm_bf16<0><<<dim3((BB * SS) / 128, HD / 64), 128, 0, stream>>>(
      hbf, wqbf, BB * SS, HD, CC, qbf, nullptr, nullptr, nullptr);
  gemm_bf16<3><<<dim3((BB * TT + 127) / 128, HD / 64), 128, 0, stream>>>(
      encbf, wkbf, BB * TT, HD, CE, kbf, kp, nullptr, nullptr);
  gemm_bf16<1><<<dim3((BB * TT + 127) / 128, HD / 64), 128, 0, stream>>>(
      encbf, wvbf, BB * TT, HD, CE, nullptr, vp, nullptr, nullptr);

  (void)hipMemsetAsync(tok_scores, 0, (size_t)BB * TT * 4, stream);
  (void)hipMemsetAsync(pos_scores, 0, (size_t)BB * SS * 4, stream);

  dim3 agrid(SS / 64, HH, BB);
  attn_tok_kernel<<<agrid, 128, 0, stream>>>(qbf, kbf, tok_scores);
  topk_mask_kernel<<<BB, 128, 0, stream>>>(tok_scores, tok_mask, TT, 12);     // ceil(77*0.15)
  attn_pos_kernel<<<agrid, 128, 0, stream>>>(qbf, kbf, tok_mask, pos_scores);
  topk_mask_kernel<<<BB, 1024, 0, stream>>>(pos_scores, pos_mask, SS, 1229);  // ceil(4096*0.3)
  mix_kv_kernel<<<(TT * HD + 255) / 256, 256, 0, stream>>>(kp, vp, tok_mask, k2bf, v2bf);
  attn2_kernel<<<agrid, 128, 0, stream>>>(qbf, k2bf, v2bf, ctx);
  inject_kernel<<<(SS * HD + 255) / 256, 256, 0, stream>>>(ctx, pos_mask, ctxbf);

  // Output projection + bias + residual (residual read in fp32 for exactness)
  gemm_bf16<2><<<dim3((BB * SS) / 128, HD / 64), 128, 0, stream>>>(
      ctxbf, wobf, BB * SS, HD, CC, nullptr, out, bo, hidden);
}
